// STGCN_85813446574118
// MI455X (gfx1250) — matrix operand order repaired and checked by reading, not compiled, not run on hardware
//
#include <hip/hip_runtime.h>
#include <hip/hip_bf16.h>

// ---------------------------------------------------------------------------
// Types for CDNA5 WMMA (wave32, gfx1250)
// ---------------------------------------------------------------------------
typedef __attribute__((ext_vector_type(16))) __bf16 v16bf;
typedef __attribute__((ext_vector_type(8)))  float  v8f;

#define DIM 128
#define LDSK 136   // padded k-stride (bf16 elems) for transposed W in LDS

// float -> bf16 bits, round-to-nearest-even
__device__ __forceinline__ unsigned int f2bf_u(float f) {
    unsigned int u = __float_as_uint(f);
    unsigned int r = u + 0x7FFFu + ((u >> 16) & 1u);
    return r >> 16;
}
__device__ __forceinline__ unsigned short f2bf(float f) {
    return (unsigned short)f2bf_u(f);
}

// ---------------------------------------------------------------------------
// Elementwise (x4 vectorized): hb[n,d] = bf16( x[n,d] + temporal_emb[t,d] )
// x = concat(user_table, item_table)
// ---------------------------------------------------------------------------
__global__ __launch_bounds__(256) void k_add_temb_bf16(
    const float* __restrict__ ut, const float* __restrict__ it,
    const float* __restrict__ temb_t, unsigned short* __restrict__ hb,
    int U, long long total4)
{
    long long g = (long long)blockIdx.x * blockDim.x + threadIdx.x;
    if (g >= total4) return;
    long long i = g * 4;
    int n = (int)(i >> 7);
    int d = (int)(i & (DIM - 1));
    const float* row = (n < U) ? ut + (size_t)n * DIM : it + (size_t)(n - U) * DIM;
    float4 x = *(const float4*)(row + d);
    float4 e = *(const float4*)(temb_t + d);
    unsigned int lo = f2bf_u(x.x + e.x) | (f2bf_u(x.y + e.y) << 16);
    unsigned int hi = f2bf_u(x.z + e.z) | (f2bf_u(x.w + e.w) << 16);
    *(uint2*)(hb + i) = make_uint2(lo, hi);
}

// ---------------------------------------------------------------------------
// Convert fp32 weights to bf16 bits
// ---------------------------------------------------------------------------
__global__ __launch_bounds__(256) void k_cvt_bf16(
    const float* __restrict__ src, unsigned short* __restrict__ dst, int n)
{
    int i = blockIdx.x * blockDim.x + threadIdx.x;
    if (i < n) dst[i] = f2bf(src[i]);
}

// ---------------------------------------------------------------------------
// Zero fp32 buffer (float4 stores)
// ---------------------------------------------------------------------------
__global__ __launch_bounds__(256) void k_zero_f32(float* __restrict__ a, long long n)
{
    long long i = ((long long)blockIdx.x * blockDim.x + threadIdx.x) * 4;
    if (i + 3 < n) {
        *(float4*)(a + i) = make_float4(0.f, 0.f, 0.f, 0.f);
    } else {
        for (; i < n; ++i) a[i] = 0.f;
    }
}

// ---------------------------------------------------------------------------
// Dense GEMM via WMMA bf16->fp32:  C[M,128] = A[M,128] @ W[128,128]
// 8 waves / block; each wave computes a 16x128 strip using 8 accumulators.
// W is staged transposed (Wt[n][k]) into LDS with padded stride.
// A-fragments are loaded before the staging barrier (overlap HBM latency);
// B-fragments are double-buffered so each ds_load has a full WMMA of slack.
// ---------------------------------------------------------------------------
__global__ __launch_bounds__(256) void k_gemm_bf16_wmma(
    const unsigned short* __restrict__ Abf,   // [M,128] bf16 bits, row-major
    const unsigned short* __restrict__ Wbf,   // [128,128] bf16 bits (k rows, n cols)
    float* __restrict__ C,                    // [M,128] fp32
    int M)
{
    __shared__ __align__(16) unsigned short lds_wt[DIM * LDSK];  // Wt[n][k]

    const int tid   = threadIdx.x;
    const int wave  = tid >> 5;
    const int lane  = tid & 31;
    const int nlo   = lane & 15;       // column-in-tile / row-in-tile
    const int khalf = lane >> 4;       // 0 or 1
    const int m_tile = blockIdx.x * 128 + wave * 16;

    // ---- Issue all A-fragment global loads first (hide behind W staging) ----
    int arow = m_tile + nlo;
    if (arow >= M) arow = M - 1;                 // clamp: stores are guarded
    const unsigned short* aptr = Abf + (size_t)arow * DIM;

    union U16 { v16bf v; uint4 q[2]; };
    U16 afrag[4];
    #pragma unroll
    for (int kt = 0; kt < 4; ++kt) {
        const int k0 = kt * 32 + khalf * 16;
        afrag[kt].q[0] = *(const uint4*)(aptr + k0);
        afrag[kt].q[1] = *(const uint4*)(aptr + k0 + 8);
    }

    // ---- Stage W transposed into LDS ----
    {
        int k  = tid >> 1;
        int c0 = (tid & 1) * 64;
        const unsigned short* src = Wbf + (size_t)k * DIM + c0;
        #pragma unroll
        for (int i = 0; i < 64; ++i)
            lds_wt[(c0 + i) * LDSK + k] = src[i];
    }
    __syncthreads();

    v8f acc[8];
    #pragma unroll
    for (int j = 0; j < 8; ++j) {
        v8f z = {};
        acc[j] = z;
    }

    // B-fragment loader: 32x16 B in contiguous k order (same as A) — lane<16: K kb..kb+15 (kb=kt*32),
    // lane>=16: K kb+16..kb+31; 32 contiguous bytes per lane from padded Wt.
    auto ldb = [&](int kt, int j) -> v16bf {
        U16 u;
        const unsigned short* bp =
            &lds_wt[(j * 16 + nlo) * LDSK + kt * 32 + khalf * 16];
        u.q[0] = *(const uint4*)(bp);
        u.q[1] = *(const uint4*)(bp + 8);
        return u.v;
    };

    // ---- Main loop: double-buffered B fragments ----
    v16bf bcur = ldb(0, 0);
    #pragma unroll
    for (int kt = 0; kt < 4; ++kt) {
        const v16bf a = afrag[kt].v;
        #pragma unroll
        for (int j = 0; j < 8; ++j) {
            const int jn  = (j + 1) & 7;
            const int ktn = (j == 7) ? kt + 1 : kt;
            v16bf bnext = bcur;
            if (!(kt == 3 && j == 7)) bnext = ldb(ktn, jn);
            acc[j] = __builtin_amdgcn_wmma_f32_16x16x32_bf16(
                false, a, false, bcur, (short)0, acc[j], false, false);
            bcur = bnext;
        }
    }

    // Store per ISA C/D layout: VGPR r -> M = (lane<16 ? r : 8+r), N = j*16 + nlo
    const int mbase = m_tile + khalf * 8;
    #pragma unroll
    for (int j = 0; j < 8; ++j) {
        const int n = j * 16 + nlo;
        #pragma unroll
        for (int r = 0; r < 8; ++r) {
            const int m = mbase + r;
            if (m < M) C[(size_t)m * DIM + n] = acc[j][r];
        }
    }
}

// ---------------------------------------------------------------------------
// COO SPMM scatter: A[dst] += val * S[src]; one wave per edge (grid-stride),
// each lane handles float4 (32 lanes x 4 = 128). fp32 atomics to L2.
// ---------------------------------------------------------------------------
__global__ __launch_bounds__(256) void k_spmm_scatter(
    const int* __restrict__ ei,      // [2,E]: dst = ei[e], src = ei[E+e]
    const float* __restrict__ av,
    const float* __restrict__ S,
    float* __restrict__ A,
    int E)
{
    const int lane = threadIdx.x & 31;
    int w  = (int)((blockIdx.x * blockDim.x + threadIdx.x) >> 5);
    const int nw = (int)((gridDim.x * blockDim.x) >> 5);

    for (int e = w; e < E; e += nw) {
        const int   dst = ei[e];
        const int   src = ei[(size_t)E + e];
        const float val = av[e];

        // prefetch next iteration's source row (global_prefetch_b8)
        if (e + nw < E) {
            int nsrc = ei[(size_t)E + e + nw];
            __builtin_prefetch(S + (size_t)nsrc * DIM + lane * 4, 0, 0);
        }

        const float4 sv = *(const float4*)(S + (size_t)src * DIM + lane * 4);
        float* ap = A + (size_t)dst * DIM + lane * 4;
        atomicAdd(ap + 0, val * sv.x);
        atomicAdd(ap + 1, val * sv.y);
        atomicAdd(ap + 2, val * sv.z);
        atomicAdd(ap + 3, val * sv.w);
    }
}

// ---------------------------------------------------------------------------
// (x4 vectorized) h = relu(A + bias); A updated in place (fp32),
// hb = bf16(h) for next GEMM
// ---------------------------------------------------------------------------
__global__ __launch_bounds__(256) void k_bias_relu(
    float* __restrict__ A, const float* __restrict__ bias,
    unsigned short* __restrict__ hb, long long total4)
{
    long long g = (long long)blockIdx.x * blockDim.x + threadIdx.x;
    if (g >= total4) return;
    long long i = g * 4;
    int d = (int)(i & (DIM - 1));
    float4 a = *(const float4*)(A + i);
    float4 b = *(const float4*)(bias + d);
    a.x = fmaxf(a.x + b.x, 0.f);
    a.y = fmaxf(a.y + b.y, 0.f);
    a.z = fmaxf(a.z + b.z, 0.f);
    a.w = fmaxf(a.w + b.w, 0.f);
    *(float4*)(A + i) = a;
    unsigned int lo = f2bf_u(a.x) | (f2bf_u(a.y) << 16);
    unsigned int hi = f2bf_u(a.z) | (f2bf_u(a.w) << 16);
    *(uint2*)(hb + i) = make_uint2(lo, hi);
}

// ---------------------------------------------------------------------------
// Gather batch rows into temporal stack: ts[b, t, 0:128]=H[uidx[b]],
// ts[b, t, 128:256]=H[U+iidx[b]]
// ---------------------------------------------------------------------------
__global__ __launch_bounds__(256) void k_gather_t(
    const float* __restrict__ H, const int* __restrict__ uidx,
    const int* __restrict__ iidx, float* __restrict__ ts,
    int t, int T, int U, int B)
{
    int i = blockIdx.x * blockDim.x + threadIdx.x;
    if (i >= B * 2 * DIM) return;
    int b = i >> 8;
    int d = i & 255;
    float v = (d < DIM) ? H[(size_t)uidx[b] * DIM + d]
                        : H[((size_t)U + iidx[b]) * DIM + (d - DIM)];
    ts[((size_t)b * T + t) * (2 * DIM) + d] = v;
}

// ---------------------------------------------------------------------------
// Fused attention + prediction head. One block (256 thr) per batch row.
// ---------------------------------------------------------------------------
__global__ __launch_bounds__(256) void k_head(
    const int* __restrict__ uidx, const int* __restrict__ iidx,
    const float* __restrict__ ut, const float* __restrict__ it,
    const float* __restrict__ ts,
    const float* __restrict__ aw1, const float* __restrict__ ab1,   // [256,128],[128]
    const float* __restrict__ aw2, const float* __restrict__ ab2,   // [128,T],[T]
    const float* __restrict__ pw1, const float* __restrict__ pb1,   // [256,128],[128]
    const float* __restrict__ pw2, const float* __restrict__ pb2,   // [128,1],[1]
    float* __restrict__ out, int T)
{
    __shared__ float c[2 * DIM];
    __shared__ float h1[DIM];
    __shared__ float attn[8];
    __shared__ float s[2 * DIM];
    __shared__ float ph[DIM];
    __shared__ float red[256];

    const int b = blockIdx.x, tid = threadIdx.x;
    const int u = uidx[b], im = iidx[b];

    c[tid] = (tid < DIM) ? ut[(size_t)u * DIM + tid]
                         : it[(size_t)im * DIM + (tid - DIM)];
    __syncthreads();

    if (tid < DIM) {
        float acc = ab1[tid];
        #pragma unroll 8
        for (int k = 0; k < 2 * DIM; ++k) acc = fmaf(c[k], aw1[k * DIM + tid], acc);
        h1[tid] = fmaxf(acc, 0.f);
    }
    __syncthreads();

    if (tid < T) {
        float acc = ab2[tid];
        for (int k = 0; k < DIM; ++k) acc = fmaf(h1[k], aw2[k * T + tid], acc);
        attn[tid] = acc;
    }
    __syncthreads();

    if (tid == 0) {
        float m = attn[0];
        for (int t = 1; t < T; ++t) m = fmaxf(m, attn[t]);
        float sum = 0.f;
        for (int t = 0; t < T; ++t) { float e = __expf(attn[t] - m); attn[t] = e; sum += e; }
        float inv = 1.f / sum;
        for (int t = 0; t < T; ++t) attn[t] *= inv;
    }
    __syncthreads();

    {
        const float* tsb = ts + (size_t)b * T * (2 * DIM);
        float acc = 0.f;
        for (int t = 0; t < T; ++t) acc = fmaf(attn[t], tsb[t * (2 * DIM) + tid], acc);
        s[tid] = acc;
    }
    __syncthreads();

    if (tid < DIM) {
        float acc = pb1[tid];
        #pragma unroll 8
        for (int k = 0; k < 2 * DIM; ++k) acc = fmaf(s[k], pw1[k * DIM + tid], acc);
        ph[tid] = fmaxf(acc, 0.f);
    }
    __syncthreads();

    red[tid] = (tid < DIM) ? ph[tid] * pw2[tid] : 0.f;
    __syncthreads();
    for (int off = 128; off >= 1; off >>= 1) {
        if (tid < off) red[tid] += red[tid + off];
        __syncthreads();
    }
    if (tid == 0) out[b] = red[0] + pb2[0];
}

// ---------------------------------------------------------------------------
// Launcher
// ---------------------------------------------------------------------------
extern "C" void kernel_launch(void* const* d_in, const int* in_sizes, int n_in,
                              void* d_out, int out_size, void* d_ws, size_t ws_size,
                              hipStream_t stream)
{
    const int*   uidx = (const int*)  d_in[0];
    const int*   iidx = (const int*)  d_in[1];
    const int*   ei   = (const int*)  d_in[2];
    const float* av   = (const float*)d_in[3];
    const float* ut   = (const float*)d_in[4];
    const float* itab = (const float*)d_in[5];
    const float* temb = (const float*)d_in[6];
    const float* gw   = (const float*)d_in[7];
    const float* gb   = (const float*)d_in[8];
    const float* aw1  = (const float*)d_in[9];
    const float* ab1  = (const float*)d_in[10];
    const float* aw2  = (const float*)d_in[11];
    const float* ab2  = (const float*)d_in[12];
    const float* pw1  = (const float*)d_in[13];
    const float* pb1  = (const float*)d_in[14];
    const float* pw2  = (const float*)d_in[15];
    const float* pb2  = (const float*)d_in[16];
    float* out = (float*)d_out;

    const int Dd = in_sizes[10];          // 128
    const int Tt = in_sizes[12];          // 3
    const int Bb = in_sizes[0];           // 4096
    const int E  = in_sizes[3];           // 3.2M
    const int U  = in_sizes[4] / Dd;      // 50000
    const int I  = in_sizes[5] / Dd;      // 50000
    const int N  = U + I;                 // 100000
    const int Ll = in_sizes[8] / Dd;      // 2
    (void)I; (void)n_in; (void)out_size; (void)ws_size;

    // Workspace partition (256B aligned)
    char* p = (char*)d_ws;
    auto carve = [&](size_t bytes) -> void* {
        void* r = (void*)p;
        p += (bytes + 255) & ~(size_t)255;
        return r;
    };
    unsigned short* wbf = (unsigned short*)carve((size_t)Ll * Dd * Dd * sizeof(unsigned short));
    unsigned short* hb  = (unsigned short*)carve((size_t)N * Dd * sizeof(unsigned short));
    float* S  = (float*)carve((size_t)N * Dd * sizeof(float));
    float* A  = (float*)carve((size_t)N * Dd * sizeof(float));
    float* ts = (float*)carve((size_t)Bb * Tt * 2 * Dd * sizeof(float));

    const long long nd  = (long long)N * Dd;
    const long long nd4 = nd / 4;
    const int blk = 256;
    const int gemm_blocks = (N + 127) / 128;

    // Convert GCN weights to bf16 once per call
    {
        int wn = Ll * Dd * Dd;
        k_cvt_bf16<<<(wn + blk - 1) / blk, blk, 0, stream>>>(gw, wbf, wn);
    }

    for (int t = 0; t < Tt; ++t) {
        k_add_temb_bf16<<<(int)((nd4 + blk - 1) / blk), blk, 0, stream>>>(
            ut, itab, temb + (size_t)t * Dd, hb, U, nd4);

        for (int l = 0; l < Ll; ++l) {
            k_gemm_bf16_wmma<<<gemm_blocks, blk, 0, stream>>>(
                hb, wbf + (size_t)l * Dd * Dd, S, N);

            k_zero_f32<<<(int)((nd4 + blk - 1) / blk) + 1, blk, 0, stream>>>(A, nd);

            k_spmm_scatter<<<8192, blk, 0, stream>>>(ei, av, S, A, E);

            k_bias_relu<<<(int)((nd4 + blk - 1) / blk), blk, 0, stream>>>(
                A, gb + (size_t)l * Dd, hb, nd4);
        }

        k_gather_t<<<(Bb * 2 * Dd + blk - 1) / blk, blk, 0, stream>>>(
            A, uidx, iidx, ts, t, Tt, U, Bb);
    }

    k_head<<<Bb, blk, 0, stream>>>(uidx, iidx, ut, itab, ts,
                                   aw1, ab1, aw2, ab2, pw1, pb1, pw2, pb2,
                                   out, Tt);
}
